// EnhancedEncoderLayer_64991445123792
// MI455X (gfx1250) — matrix operand order repaired and checked by reading, not compiled, not run on hardware
//
#include <hip/hip_runtime.h>
#include <stdint.h>

typedef __attribute__((ext_vector_type(16))) __bf16 v16bf;
typedef __attribute__((ext_vector_type(8)))  float  v8f;
typedef __attribute__((ext_vector_type(4)))  unsigned int uint32x4;
typedef __attribute__((ext_vector_type(8)))  int int32x8;
typedef __attribute__((ext_vector_type(4)))  int int32x4;

#if defined(__HIP_DEVICE_COMPILE__) && __has_builtin(__builtin_amdgcn_tensor_load_to_lds) && __has_builtin(__builtin_amdgcn_s_wait_tensorcnt)
#define USE_TDM 1
#else
#define USE_TDM 0
#endif

__device__ __forceinline__ unsigned short f2bf(float f) {
  unsigned u = __float_as_uint(f);
  unsigned r = u + 0x7FFFu + ((u >> 16) & 1u);   // round-to-nearest-even
  return (unsigned short)(r >> 16);
}

union FragB16 { uint4 q[2]; v16bf v; };

#if USE_TDM
// Issue a TDM load of a 128-row x 32-element (bf16) strided tile into LDS.
// D# per CDNA5 ISA 8.3/8.4: the "tensor" is declared to be exactly the tile
// (tensor_dim = tile_dim) so no OOB clipping ever applies.
__device__ __forceinline__ void tdm_load_tile_128x32(const unsigned short* gsrc,
                                                     unsigned ldsAddr, int ldElems)
{
  unsigned long long ga = (unsigned long long)(uintptr_t)gsrc;
  uint32x4 g0;
  g0[0] = 1u;                                              // count=1 (valid), user mode
  g0[1] = ldsAddr;                                         // lds_addr (bytes)
  g0[2] = (unsigned)(ga & 0xffffffffull);                  // global_addr[31:0]
  g0[3] = (unsigned)((ga >> 32) & 0x01ffffffull) | 0x80000000u; // addr[56:32] | type=2
  const unsigned td0 = 32u, td1 = 128u;                    // tensor dims (elements)
  const unsigned tl0 = 32u, tl1 = 128u;                    // tile dims
  unsigned long long st = (unsigned long long)(unsigned)ldElems; // dim0 stride (elements)
  int32x8 g1;
  g1[0] = (int)(1u << 16);                                 // workgroup_mask=0, data_size=1 (2B)
  g1[1] = (int)((td0 & 0xffffu) << 16);                    // tensor_dim0[15:0]
  g1[2] = (int)(((td0 >> 16) & 0xffffu) | ((td1 & 0xffffu) << 16)); // dim0[31:16], dim1[15:0]
  g1[3] = (int)(((td1 >> 16) & 0xffffu) | (tl0 << 16));    // dim1[31:16], tile_dim0
  g1[4] = (int)(tl1 & 0xffffu);                            // tile_dim1 (tile_dim2=0)
  g1[5] = (int)(st & 0xffffffffull);                       // tensor_dim0_stride[31:0]
  g1[6] = (int)((st >> 32) & 0xffffull);                   // stride[47:32]; dim1_stride=0
  g1[7] = 0;
  int32x4 z4 = {0, 0, 0, 0};
#if __clang_major__ >= 23
  int32x8 z8 = {0, 0, 0, 0, 0, 0, 0, 0};
  __builtin_amdgcn_tensor_load_to_lds(g0, g1, z4, z4, z8, 0);
#else
  __builtin_amdgcn_tensor_load_to_lds(g0, g1, z4, z4, 0);
#endif
}
#endif

// ---------------------------------------------------------------------------
// Generic bf16 WMMA GEMM: C = epilogue( kscale * (A @ B) )
//   A: bf16 row-major [M,K], per-z base = (z/aBI)*aSO + (z%aBI)*aSI
//   B: BMODE=0: bf16 row-major [K,N] (ldb = k-row stride)
//      BMODE=1: bf16 row-major [N,K] (ldb = n-row stride, contiguous k) -> B^T
//   Epilogue: +bias[n], ACT (1=relu, 2=sigmoid), *ew[off], res[off]+scale*v,
//             output f32 or bf16.
// Block: 256 thr (8 waves), tile 128x128, BK=32, double-buffered LDS.
// TDM (tensor_load_to_lds) prefetches the next K-tile while WMMAs consume the
// current one; TENSORcnt gates the handoff.
// ---------------------------------------------------------------------------
template<int BMODE, int OUTBF16, int ACT, int HASBIAS, int EW, int RES>
__global__ __launch_bounds__(256)
void gemm_wmma(const unsigned short* __restrict__ Ag, long long aSO, long long aSI, int aBI, int lda,
               const unsigned short* __restrict__ Bg, long long bSO, long long bSI, int bBI, int ldb,
               const float* __restrict__ bias,
               const float* __restrict__ ewp,
               const float* __restrict__ resp, const float* __restrict__ scalep,
               float* __restrict__ Cf, unsigned short* __restrict__ Cb,
               long long cSO, long long cSI, int cBI, int ldc,
               int M, int N, int K, float kscale)
{
  __shared__ __align__(16) unsigned short As[2][128 * 32];   // [m][k]
  __shared__ __align__(16) unsigned short Bs[2][128 * 32];   // [n][k]
  const int tid  = threadIdx.x;
  const int wave = tid >> 5, lane = tid & 31;
  const int wr = wave >> 1, wc = wave & 1;                // 4x2 wave grid
  const int lane16 = lane & 15, laneH = lane >> 4;
  const int z  = blockIdx.z;
  const int m0 = blockIdx.y * 128, n0 = blockIdx.x * 128;
  const unsigned short* A  = Ag + (long long)(z / aBI) * aSO + (long long)(z % aBI) * aSI;
  const unsigned short* Bp = Bg + (long long)(z / bBI) * bSO + (long long)(z % bBI) * bSI;
  const long long cbase    = (long long)(z / cBI) * cSO + (long long)(z % cBI) * cSI;

  const v8f vzero = {0.f, 0.f, 0.f, 0.f, 0.f, 0.f, 0.f, 0.f};
  v8f acc[2][4];
#pragma unroll
  for (int i = 0; i < 2; ++i)
#pragma unroll
    for (int j = 0; j < 4; ++j) acc[i][j] = vzero;

#if USE_TDM
  const unsigned ldsA0 = (unsigned)(unsigned long long)(uintptr_t)&As[0][0];
  const unsigned ldsA1 = (unsigned)(unsigned long long)(uintptr_t)&As[1][0];
  const unsigned ldsB0 = (unsigned)(unsigned long long)(uintptr_t)&Bs[0][0];
  const unsigned ldsB1 = (unsigned)(unsigned long long)(uintptr_t)&Bs[1][0];
  constexpr short NPRE = (BMODE == 1) ? 2 : 1;   // tensor ops in flight per tile
  if (wave == 0) {                                // prologue: DMA first tile -> buf0
    tdm_load_tile_128x32(A + (long long)m0 * lda, ldsA0, lda);
    if (BMODE == 1)
      tdm_load_tile_128x32(Bp + (long long)n0 * ldb, ldsB0, ldb);
  }
#endif

  for (int k0 = 0; k0 < K; k0 += 32) {
    const int buf = (k0 >> 5) & 1;
#if USE_TDM
    const int nbuf = buf ^ 1;
    if (wave == 0) {
      if (k0 + 32 < K) {                          // prefetch next tile into other buf
        tdm_load_tile_128x32(A + (long long)m0 * lda + k0 + 32,
                             nbuf ? ldsA1 : ldsA0, lda);
        if (BMODE == 1)
          tdm_load_tile_128x32(Bp + (long long)n0 * ldb + k0 + 32,
                               nbuf ? ldsB1 : ldsB0, ldb);
        __builtin_amdgcn_s_wait_tensorcnt(NPRE);  // current tile complete
      } else {
        __builtin_amdgcn_s_wait_tensorcnt(0);
      }
    }
    if (BMODE == 0) {                             // transpose-scatter B (TDM can't transpose)
#pragma unroll
      for (int r = 0; r < 2; ++r) {
        int u = tid + (r << 8);
        int kk = u >> 4, c8 = (u & 15) << 3;
        uint4 val = {0u, 0u, 0u, 0u};
        if (n0 + c8 < N)
          val = *reinterpret_cast<const uint4*>(Bp + (long long)(k0 + kk) * ldb + n0 + c8);
        unsigned short tmp[8];
        __builtin_memcpy(tmp, &val, 16);
#pragma unroll
        for (int j = 0; j < 8; ++j) Bs[buf][(c8 + j) * 32 + kk] = tmp[j];
      }
    }
#else
    // ---- fallback: per-thread global->LDS staging ----
#pragma unroll
    for (int r = 0; r < 2; ++r) {
      int u = tid + (r << 8);
      int row = u >> 2, c8 = (u & 3) << 3;
      const uint4* src = reinterpret_cast<const uint4*>(A + (long long)(m0 + row) * lda + k0 + c8);
      *reinterpret_cast<uint4*>(&As[buf][row * 32 + c8]) = *src;
    }
    if (BMODE == 0) {
#pragma unroll
      for (int r = 0; r < 2; ++r) {
        int u = tid + (r << 8);
        int kk = u >> 4, c8 = (u & 15) << 3;
        uint4 val = {0u, 0u, 0u, 0u};
        if (n0 + c8 < N)
          val = *reinterpret_cast<const uint4*>(Bp + (long long)(k0 + kk) * ldb + n0 + c8);
        unsigned short tmp[8];
        __builtin_memcpy(tmp, &val, 16);
#pragma unroll
        for (int j = 0; j < 8; ++j) Bs[buf][(c8 + j) * 32 + kk] = tmp[j];
      }
    } else {
#pragma unroll
      for (int r = 0; r < 2; ++r) {
        int u = tid + (r << 8);
        int nn = u >> 2, c8 = (u & 3) << 3;
        uint4 val = {0u, 0u, 0u, 0u};
        if (n0 + nn < N)
          val = *reinterpret_cast<const uint4*>(Bp + (long long)(n0 + nn) * ldb + k0 + c8);
        *reinterpret_cast<uint4*>(&Bs[buf][nn * 32 + c8]) = val;
      }
    }
    if (k0 + 32 < K)                              // CDNA5 global_prefetch_b8
      __builtin_prefetch(A + (long long)(m0 + (tid >> 2)) * lda + k0 + 32, 0, 1);
#endif
    __syncthreads();                              // publish LDS tile to all waves

    // ---- fragments per ISA 7.12.2 layouts ----
    FragB16 af[2], bfr[4];
#pragma unroll
    for (int mt = 0; mt < 2; ++mt) {
      int m = (wr * 32 + mt * 16 + lane16) * 32;
      af[mt].q[0] = *reinterpret_cast<const uint4*>(&As[buf][m + laneH * 8]);       // K = 8h..
      af[mt].q[1] = *reinterpret_cast<const uint4*>(&As[buf][m + 16 + laneH * 8]);  // K = 16+8h..
    }
#pragma unroll
    for (int nt = 0; nt < 4; ++nt) {
      int n = (wc * 64 + nt * 16 + lane16) * 32;
      bfr[nt].q[0] = *reinterpret_cast<const uint4*>(&Bs[buf][n + laneH * 8]);      // K = 8h..
      bfr[nt].q[1] = *reinterpret_cast<const uint4*>(&Bs[buf][n + 16 + laneH * 8]); // K = 16+8h..
    }
#pragma unroll
    for (int mt = 0; mt < 2; ++mt)
#pragma unroll
      for (int nt = 0; nt < 4; ++nt)
        acc[mt][nt] = __builtin_amdgcn_wmma_f32_16x16x32_bf16(
            false, af[mt].v, false, bfr[nt].v, (short)0, acc[mt][nt], false, false);

    __syncthreads();                              // all reads done before buf reuse
  }

  const float rscale = RES ? scalep[0] : 1.0f;
#pragma unroll
  for (int mt = 0; mt < 2; ++mt) {
#pragma unroll
    for (int nt = 0; nt < 4; ++nt) {
      int n = n0 + wc * 64 + nt * 16 + lane16;
      if (n < N) {
        float bvv = HASBIAS ? bias[n] : 0.0f;
#pragma unroll
        for (int j = 0; j < 8; ++j) {
          int m = m0 + wr * 32 + mt * 16 + laneH * 8 + j;   // C layout: M = j + 8*half
          long long off = cbase + (long long)m * ldc + n;
          float v = acc[mt][nt][j] * kscale + bvv;
          if (ACT == 1) v = fmaxf(v, 0.0f);
          if (ACT == 2) v = 1.0f / (1.0f + __expf(-v));
          if (EW)  v *= ewp[off];
          if (RES) v = resp[off] + rscale * v;
          if (OUTBF16) Cb[off] = f2bf(v);
          else         Cf[off] = v;
        }
      }
    }
  }
}

// ---------------------------------------------------------------------------
// Row softmax over L=1024 cols; normalizes f32 in place (d_out attn region)
// and writes a bf16 copy for the attn@V WMMA GEMM.
// ---------------------------------------------------------------------------
__global__ __launch_bounds__(256)
void softmax_rows(float* __restrict__ attn, unsigned short* __restrict__ attnb, int Lc)
{
  __shared__ float red[256];
  const int tid = threadIdx.x;
  const long long row = blockIdx.x;
  float* p = attn + row * Lc;
  float4 v = reinterpret_cast<float4*>(p)[tid];
  float mx = fmaxf(fmaxf(v.x, v.y), fmaxf(v.z, v.w));
  red[tid] = mx; __syncthreads();
  for (int s = 128; s > 0; s >>= 1) { if (tid < s) red[tid] = fmaxf(red[tid], red[tid + s]); __syncthreads(); }
  mx = red[0]; __syncthreads();
  v.x = __expf(v.x - mx); v.y = __expf(v.y - mx); v.z = __expf(v.z - mx); v.w = __expf(v.w - mx);
  red[tid] = v.x + v.y + v.z + v.w; __syncthreads();
  for (int s = 128; s > 0; s >>= 1) { if (tid < s) red[tid] += red[tid + s]; __syncthreads(); }
  float inv = 1.0f / red[0];
  v.x *= inv; v.y *= inv; v.z *= inv; v.w *= inv;
  reinterpret_cast<float4*>(p)[tid] = v;
  ushort4 o; o.x = f2bf(v.x); o.y = f2bf(v.y); o.z = f2bf(v.z); o.w = f2bf(v.w);
  reinterpret_cast<ushort4*>(attnb + row * Lc)[tid] = o;
}

// ---------------------------------------------------------------------------
// Per-batch kernel-size predictor: mean over L -> MLP -> k -> masked softmax
// of trend weights. One 256-thread block per batch (D=512, D/2=256).
// ---------------------------------------------------------------------------
__global__ __launch_bounds__(256)
void kpredict(const float* __restrict__ xin, const float* __restrict__ w1, const float* __restrict__ b1,
              const float* __restrict__ w2, const float* __restrict__ b2,
              const float* __restrict__ tw, float* __restrict__ wout, int* __restrict__ khout,
              int Lr, int Dd)
{
  __shared__ float xg[512];
  __shared__ float red[256];
  const int tid = threadIdx.x;
  const int b = blockIdx.x;
  const float invL = 1.0f / (float)Lr;
  for (int r = 0; r < 2; ++r) {
    int d = tid + (r << 8);
    float acc = 0.0f;
    const float* src = xin + (long long)b * Lr * Dd + d;
    for (int t = 0; t < Lr; ++t) acc += src[(long long)t * Dd];
    xg[d] = acc * invL;
  }
  __syncthreads();
  float h = b1[tid];
  for (int d = 0; d < 512; ++d) h += xg[d] * w1[d * 256 + tid];
  h = fmaxf(h, 0.0f);
  red[tid] = h * w2[tid];
  __syncthreads();
  for (int s = 128; s > 0; s >>= 1) { if (tid < s) red[tid] += red[tid + s]; __syncthreads(); }
  if (tid == 0) {
    float sg = 1.0f / (1.0f + __expf(-(red[0] + b2[0])));
    int k = (int)roundf(sg * 45.0f + 5.0f);     // MAX_K-5 = 45
    k = k < 5 ? 5 : (k > 50 ? 50 : k);          // min(MAX_K, L/3) = 50
    khout[b] = k >> 1;
    float m = -3.0e38f;
    for (int i = 0; i < k; ++i) m = fmaxf(m, tw[i]);
    float e[50]; float sum = 0.0f;
    for (int i = 0; i < k; ++i) { e[i] = __expf(tw[i] - m); sum += e[i]; }
    float inv = 1.0f / sum;
    for (int i = 0; i < 64; ++i) wout[b * 64 + i] = (i < k) ? e[i] * inv : 0.0f;
  }
}

// ---------------------------------------------------------------------------
// Seasonal = x - replicate-padded 50-tap depthwise conv (weights per batch).
// One block per (b,t); re-reads hit the 192MB L2 (input tensor is 16MB).
// ---------------------------------------------------------------------------
__global__ __launch_bounds__(256)
void decomp_seasonal(const float* __restrict__ xin, const float* __restrict__ wdec,
                     const int* __restrict__ kh, float* __restrict__ out, int Lr, int Dd)
{
  __shared__ float wsm[64];
  __shared__ int khs;
  const int tid = threadIdx.x;
  const int b = blockIdx.x / Lr;
  const int t = blockIdx.x % Lr;
  if (tid < 64) wsm[tid] = wdec[b * 64 + tid];
  if (tid == 0) khs = kh[b];
  __syncthreads();
  const long long base = (long long)b * Lr * Dd;
  const int k2 = khs;
  for (int r = 0; r < 2; ++r) {
    int d = tid + (r << 8);
    float acc = 0.0f;
#pragma unroll
    for (int i = 0; i < 50; ++i) {
      int s = t - k2 + i;
      s = s < 0 ? 0 : (s > Lr - 1 ? Lr - 1 : s);
      acc += wsm[i] * xin[base + (long long)s * Dd + d];
    }
    long long o = base + (long long)t * Dd + d;
    out[o] = xin[o] - acc;
  }
}

// ---------------------------------------------------------------------------
__global__ void f32_to_bf16_k(const float* __restrict__ in, unsigned short* __restrict__ out, long long n)
{
  long long i = (((long long)blockIdx.x * blockDim.x) + threadIdx.x) * 4;
  if (i + 3 < n) {
    float4 v = *reinterpret_cast<const float4*>(in + i);
    ushort4 o; o.x = f2bf(v.x); o.y = f2bf(v.y); o.z = f2bf(v.z); o.w = f2bf(v.w);
    *reinterpret_cast<ushort4*>(out + i) = o;
  }
}

// out is row-major [K,N]; trans: in is [N,K] row-major (weight used as x @ W^T)
__global__ void convert_weight_k(const float* __restrict__ in, unsigned short* __restrict__ out,
                                 int K, int N, int trans)
{
  long long i = (long long)blockIdx.x * blockDim.x + threadIdx.x;
  if (i < (long long)K * N) {
    int k = (int)(i / N), n = (int)(i % N);
    float v = trans ? in[(long long)n * K + k] : in[i];
    out[i] = f2bf(v);
  }
}

// ---------------------------------------------------------------------------
extern "C" void kernel_launch(void* const* d_in, const int* in_sizes, int n_in,
                              void* d_out, int out_size, void* d_ws, size_t ws_size,
                              hipStream_t stream)
{
  (void)in_sizes; (void)n_in; (void)out_size; (void)ws_size;
  constexpr int Bv = 8, Lv = 1024, Dv = 512, Hv = 8, Hdv = 64, FF = 2048;
  const long long BLD = (long long)Bv * Lv * Dv;

  const float* x    = (const float*)d_in[0];
  const float* Wq   = (const float*)d_in[1];  const float* bq   = (const float*)d_in[2];
  const float* Wk   = (const float*)d_in[3];  const float* bk   = (const float*)d_in[4];
  const float* Wv   = (const float*)d_in[5];  const float* bvp  = (const float*)d_in[6];
  const float* Wo   = (const float*)d_in[7];  const float* bo   = (const float*)d_in[8];
  const float* gw   = (const float*)d_in[9];  const float* gbv  = (const float*)d_in[10];
  const float* c1w  = (const float*)d_in[11];
  const float* c2w  = (const float*)d_in[12];
  const float* tw1  = (const float*)d_in[13];
  const float* k1w1 = (const float*)d_in[14]; const float* k1b1 = (const float*)d_in[15];
  const float* k1w2 = (const float*)d_in[16]; const float* k1b2 = (const float*)d_in[17];
  const float* tw2  = (const float*)d_in[18];
  const float* k2w1 = (const float*)d_in[19]; const float* k2b1 = (const float*)d_in[20];
  const float* k2w2 = (const float*)d_in[21]; const float* k2b2 = (const float*)d_in[22];
  const float* ascale = (const float*)d_in[23];
  const float* fscale = (const float*)d_in[24];

  float* outRes  = (float*)d_out;                 // seasonal [B,L,D]
  float* outAttn = (float*)d_out + BLD;           // attn [B,H,L,L]

  char* wsp = (char*)d_ws;
  auto alloc = [&](size_t bytes) -> char* {
    char* p = wsp; wsp += (bytes + 255) & ~(size_t)255; return p;
  };
  float* xa = (float*)alloc(BLD * 4);
  float* s1 = (float*)alloc(BLD * 4);
  float* x2 = (float*)alloc(BLD * 4);
  unsigned short* xb   = (unsigned short*)alloc(BLD * 2);
  unsigned short* qb   = (unsigned short*)alloc(BLD * 2);
  unsigned short* kbb  = (unsigned short*)alloc(BLD * 2);
  unsigned short* vb   = (unsigned short*)alloc(BLD * 2);
  unsigned short* ctxb = (unsigned short*)alloc(BLD * 2);
  unsigned short* s1b  = (unsigned short*)alloc(BLD * 2);
  unsigned short* gbb  = (unsigned short*)alloc(BLD * 2);
  unsigned short* y1b  = (unsigned short*)alloc((long long)Bv * Lv * FF * 2);
  unsigned short* attnb= (unsigned short*)alloc((long long)Bv * Hv * Lv * Lv * 2);
  unsigned short* wqb  = (unsigned short*)alloc((size_t)Dv * Dv * 2);
  unsigned short* wkb  = (unsigned short*)alloc((size_t)Dv * Dv * 2);
  unsigned short* wvb  = (unsigned short*)alloc((size_t)Dv * Dv * 2);
  unsigned short* wob  = (unsigned short*)alloc((size_t)Dv * Dv * 2);
  unsigned short* gwTb = (unsigned short*)alloc((size_t)Dv * Dv * 2);
  unsigned short* c1Tb = (unsigned short*)alloc((size_t)Dv * FF * 2);
  unsigned short* c2Tb = (unsigned short*)alloc((size_t)FF * Dv * 2);
  float* wdec1 = (float*)alloc(Bv * 64 * 4);
  float* wdec2 = (float*)alloc(Bv * 64 * 4);
  int*   kh1   = (int*)alloc(Bv * 4);
  int*   kh2   = (int*)alloc(Bv * 4);

  const dim3 blk(256);
  const long long LD = (long long)Lv * Dv, LL = (long long)Lv * Lv;

  // ---- precision conversion ----
  f32_to_bf16_k<<<(int)(BLD / 1024), blk, 0, stream>>>(x, xb, BLD);
  convert_weight_k<<<(Dv * Dv + 255) / 256, blk, 0, stream>>>(Wq,  wqb,  Dv, Dv, 0);
  convert_weight_k<<<(Dv * Dv + 255) / 256, blk, 0, stream>>>(Wk,  wkb,  Dv, Dv, 0);
  convert_weight_k<<<(Dv * Dv + 255) / 256, blk, 0, stream>>>(Wv,  wvb,  Dv, Dv, 0);
  convert_weight_k<<<(Dv * Dv + 255) / 256, blk, 0, stream>>>(Wo,  wob,  Dv, Dv, 0);
  convert_weight_k<<<(Dv * Dv + 255) / 256, blk, 0, stream>>>(gw,  gwTb, Dv, Dv, 1);
  convert_weight_k<<<(Dv * FF + 255) / 256, blk, 0, stream>>>(c1w, c1Tb, Dv, FF, 1);
  convert_weight_k<<<(FF * Dv + 255) / 256, blk, 0, stream>>>(c2w, c2Tb, FF, Dv, 1);

  // ---- Q/K/V projections: [8192,512] x [512,512] + bias -> bf16 ----
  gemm_wmma<0,1,0,1,0,0><<<dim3(Dv/128, (Bv*Lv)/128, 1), blk, 0, stream>>>(
      xb, 0, 0, 1, Dv, wqb, 0, 0, 1, Dv, bq, nullptr, nullptr, nullptr,
      nullptr, qb, 0, 0, 1, Dv, Bv*Lv, Dv, Dv, 1.0f);
  gemm_wmma<0,1,0,1,0,0><<<dim3(Dv/128, (Bv*Lv)/128, 1), blk, 0, stream>>>(
      xb, 0, 0, 1, Dv, wkb, 0, 0, 1, Dv, bk, nullptr, nullptr, nullptr,
      nullptr, kbb, 0, 0, 1, Dv, Bv*Lv, Dv, Dv, 1.0f);
  gemm_wmma<0,1,0,1,0,0><<<dim3(Dv/128, (Bv*Lv)/128, 1), blk, 0, stream>>>(
      xb, 0, 0, 1, Dv, wvb, 0, 0, 1, Dv, bvp, nullptr, nullptr, nullptr,
      nullptr, vb, 0, 0, 1, Dv, Bv*Lv, Dv, Dv, 1.0f);

  // ---- scores = q @ k^T / sqrt(Hd), per (b,h), f32 -> d_out attn region ----
  gemm_wmma<1,0,0,0,0,0><<<dim3(Lv/128, Lv/128, Bv*Hv), blk, 0, stream>>>(
      qb,  LD, Hdv, Hv, Dv,
      kbb, LD, Hdv, Hv, Dv,
      nullptr, nullptr, nullptr, nullptr,
      outAttn, nullptr, LL, 0, 1, Lv,
      Lv, Lv, Hdv, 0.125f);

  // ---- softmax (in-place f32) + bf16 copy ----
  softmax_rows<<<Bv * Hv * Lv, blk, 0, stream>>>(outAttn, attnb, Lv);

  // ---- ctx = attn @ v : [1024,1024] x [1024,64] per head -> bf16 [B,L,D] ----
  gemm_wmma<0,1,0,0,0,0><<<dim3(1, Lv/128, Bv*Hv), blk, 0, stream>>>(
      attnb, LL, 0, 1, Lv,
      vb, LD, Hdv, Hv, Dv,
      nullptr, nullptr, nullptr, nullptr,
      nullptr, ctxb, LD, Hdv, Hv, Dv,
      Lv, Hdv, Lv, 1.0f);

  // ---- xa = x + attn_scale * (ctx @ Wo + bo) ----
  gemm_wmma<0,0,0,1,0,1><<<dim3(Dv/128, (Bv*Lv)/128, 1), blk, 0, stream>>>(
      ctxb, 0, 0, 1, Dv, wob, 0, 0, 1, Dv, bo, nullptr, x, ascale,
      xa, nullptr, 0, 0, 1, Dv, Bv*Lv, Dv, Dv, 1.0f);

  // ---- decomp 1 ----
  kpredict<<<Bv, blk, 0, stream>>>(xa, k1w1, k1b1, k1w2, k1b2, tw1, wdec1, kh1, Lv, Dv);
  decomp_seasonal<<<Bv * Lv, blk, 0, stream>>>(xa, wdec1, kh1, s1, Lv, Dv);
  f32_to_bf16_k<<<(int)(BLD / 1024), blk, 0, stream>>>(s1, s1b, BLD);

  // ---- gated FFN: g = sigmoid(s1 @ gw^T + gb) * s1 -> bf16 ----
  gemm_wmma<0,1,2,1,1,0><<<dim3(Dv/128, (Bv*Lv)/128, 1), blk, 0, stream>>>(
      s1b, 0, 0, 1, Dv, gwTb, 0, 0, 1, Dv, gbv, s1, nullptr, nullptr,
      nullptr, gbb, 0, 0, 1, Dv, Bv*Lv, Dv, Dv, 1.0f);
  // y1 = relu(g @ c1^T) -> bf16 [B,L,FF]
  gemm_wmma<0,1,1,0,0,0><<<dim3(FF/128, (Bv*Lv)/128, 1), blk, 0, stream>>>(
      gbb, 0, 0, 1, Dv, c1Tb, 0, 0, 1, FF, nullptr, nullptr, nullptr, nullptr,
      nullptr, y1b, 0, 0, 1, FF, Bv*Lv, FF, Dv, 1.0f);
  // x2 = s1 + ffn_scale * (y1 @ c2^T)
  gemm_wmma<0,0,0,0,0,1><<<dim3(Dv/128, (Bv*Lv)/128, 1), blk, 0, stream>>>(
      y1b, 0, 0, 1, FF, c2Tb, 0, 0, 1, Dv, nullptr, nullptr, s1, fscale,
      x2, nullptr, 0, 0, 1, Dv, Bv*Lv, Dv, FF, 1.0f);

  // ---- decomp 2 -> final seasonal output ----
  kpredict<<<Bv, blk, 0, stream>>>(x2, k2w1, k2b1, k2w2, k2b2, tw2, wdec2, kh2, Lv, Dv);
  decomp_seasonal<<<Bv * Lv, blk, 0, stream>>>(x2, wdec2, kh2, outRes, Lv, Dv);
}
